// Attention_56032143344222
// MI455X (gfx1250) — matrix operand order repaired and checked by reading, not compiled, not run on hardware
//
#include <hip/hip_runtime.h>
#include <hip/hip_bf16.h>
#include <cstdint>

// Problem dims (fixed by the reference)
#define B_  16
#define LQ_ 2048
#define LK_ 2048
#define H_  1024

// CDNA5 WMMA vector types
typedef __bf16 bf16_t;
typedef __attribute__((ext_vector_type(16))) __bf16 v16bf;
typedef __attribute__((ext_vector_type(8)))  __bf16 v8bf;
typedef __attribute__((ext_vector_type(8)))  float  v8f;

// GEMM tiling: 128x128 block tile, 8 waves (4M x 2N), wave tile 32x64,
// K-step 32 == WMMA K. LDS rows padded to 80B (40 bf16) for bank spread.
// Double-buffered LDS ping-pong to overlap global loads with WMMA.
#define MT   128
#define NTL  128
#define KT   32
#define ROWE 40

enum { EPI_STORE = 0, EPI_TANH = 1 };

// D[m,n] = sum_k A[m,k] * B[n,k]   (NT; BNN=true means B source is [K][N]).
// ACONCAT: A is [A0 | A1] concatenated along K (each width K/2, same LDA).
// All dims/strides compile-time so stores use immediate offsets.
template<int EPI, bool BNN, bool ACONCAT, int K, int LDA, int LDB, int LDD>
__global__ __launch_bounds__(256) void gemm_bf16_nt(
    const float* __restrict__ A0, const float* __restrict__ A1,
    const float* __restrict__ Bsrc, const float* __restrict__ bias,
    float* __restrict__ D,
    long long sA, long long sB, long long sD)
{
  __shared__ __align__(16) bf16_t lsA[2][MT  * ROWE];
  __shared__ __align__(16) bf16_t lsB[2][NTL * ROWE];

  const int tid  = threadIdx.x;
  const int lane = tid & 31;
  const int lo   = lane & 15;
  const int hi   = lane >> 4;
  const int w    = tid >> 5;   // wave id 0..7
  const int wm   = w >> 1;     // 0..3
  const int wn   = w & 1;      // 0..1

  const int bz = blockIdx.z;
  const int m0 = blockIdx.y * MT;
  const int n0 = blockIdx.x * NTL;

  const float* Ab  = A0 + (long long)bz * sA;
  const float* A1b = ACONCAT ? (A1 + (long long)bz * sA) : nullptr;
  const float* Bb  = Bsrc + (long long)bz * sB;
  float*       Db  = D + (long long)bz * sD;

  // Staging decode: slot s = tid + i*256 -> row = s>>2 (0..127), kchunk = (s&3)*8
  const int sRow = tid >> 2;        // + i*64
  const int sKc  = (tid & 3) * 8;
  // BNN decode: slot s = tid + i*256 -> kk = s>>5 (0..31), n4 = (s&31)*4
  const int sKk  = tid >> 5;        // + i*8
  const int sN4  = (tid & 31) * 4;

  float4 rA[2][2], rBt[2][2], rBn[4];

  // ---- issue global loads for tile starting at k0 into registers ----
  auto load_tile = [&](int k0) {
    const float* as; int ak;
    if constexpr (ACONCAT) {
      constexpr int half = K / 2;
      as = (k0 < half) ? Ab : A1b;
      ak = (k0 < half) ? k0 : k0 - half;
    } else { as = Ab; ak = k0; }
    #pragma unroll
    for (int i = 0; i < 2; ++i) {
      const float* src = as + (m0 + sRow + i * 64) * LDA + ak + sKc;
      rA[i][0] = *(const float4*)src;
      rA[i][1] = *(const float4*)(src + 4);
    }
    if constexpr (!BNN) {
      #pragma unroll
      for (int i = 0; i < 2; ++i) {
        const float* src = Bb + (n0 + sRow + i * 64) * LDB + k0 + sKc;
        rBt[i][0] = *(const float4*)src;
        rBt[i][1] = *(const float4*)(src + 4);
      }
    } else {
      #pragma unroll
      for (int i = 0; i < 4; ++i)
        rBn[i] = *(const float4*)(Bb + (k0 + sKk + i * 8) * LDB + n0 + sN4);
    }
  };

  // ---- convert f32 -> bf16 (hw cvt, RNE) and store staged regs to LDS ----
  auto store_tile = [&](int buf) {
    #pragma unroll
    for (int i = 0; i < 2; ++i) {
      bf16_t* dst = &lsA[buf][(sRow + i * 64) * ROWE + sKc];
      dst[0] = (bf16_t)rA[i][0].x; dst[1] = (bf16_t)rA[i][0].y;
      dst[2] = (bf16_t)rA[i][0].z; dst[3] = (bf16_t)rA[i][0].w;
      dst[4] = (bf16_t)rA[i][1].x; dst[5] = (bf16_t)rA[i][1].y;
      dst[6] = (bf16_t)rA[i][1].z; dst[7] = (bf16_t)rA[i][1].w;
    }
    if constexpr (!BNN) {
      #pragma unroll
      for (int i = 0; i < 2; ++i) {
        bf16_t* dst = &lsB[buf][(sRow + i * 64) * ROWE + sKc];
        dst[0] = (bf16_t)rBt[i][0].x; dst[1] = (bf16_t)rBt[i][0].y;
        dst[2] = (bf16_t)rBt[i][0].z; dst[3] = (bf16_t)rBt[i][0].w;
        dst[4] = (bf16_t)rBt[i][1].x; dst[5] = (bf16_t)rBt[i][1].y;
        dst[6] = (bf16_t)rBt[i][1].z; dst[7] = (bf16_t)rBt[i][1].w;
      }
    } else {
      #pragma unroll
      for (int i = 0; i < 4; ++i) {
        const int kk = sKk + i * 8;
        lsB[buf][(sN4 + 0) * ROWE + kk] = (bf16_t)rBn[i].x;
        lsB[buf][(sN4 + 1) * ROWE + kk] = (bf16_t)rBn[i].y;
        lsB[buf][(sN4 + 2) * ROWE + kk] = (bf16_t)rBn[i].z;
        lsB[buf][(sN4 + 3) * ROWE + kk] = (bf16_t)rBn[i].w;
      }
    }
  };

  v8f acc[2][4];
  #pragma unroll
  for (int i = 0; i < 2; ++i)
    #pragma unroll
    for (int j = 0; j < 4; ++j) {
      v8f z = {0.f, 0.f, 0.f, 0.f, 0.f, 0.f, 0.f, 0.f};
      acc[i][j] = z;
    }

  // ---- fragment loads (documented CDNA5 VGPR layouts) + 8 WMMAs ----
  auto compute = [&](int buf) {
    union FragU { v16bf v; v8bf h[2]; };
    FragU af[2], bfr[4];
    #pragma unroll
    for (int i = 0; i < 2; ++i) {
      const bf16_t* base = &lsA[buf][(wm * 32 + i * 16 + lo) * ROWE];
      af[i].h[0] = *(const v8bf*)(base + hi * 8);       // K = 8*hi .. +8
      af[i].h[1] = *(const v8bf*)(base + 16 + hi * 8);  // K = 16+8*hi .. +8
    }
    #pragma unroll
    for (int j = 0; j < 4; ++j) {
      const bf16_t* base = &lsB[buf][(wn * 64 + j * 16 + lo) * ROWE];
      bfr[j].h[0] = *(const v8bf*)(base + hi * 8);       // K = 8*hi .. +8
      bfr[j].h[1] = *(const v8bf*)(base + 16 + hi * 8);  // K = 16+8*hi .. +8
    }
    #pragma unroll
    for (int i = 0; i < 2; ++i)
      #pragma unroll
      for (int j = 0; j < 4; ++j)
        acc[i][j] = __builtin_amdgcn_wmma_f32_16x16x32_bf16(
            false, af[i].v, false, bfr[j].v, (short)0, acc[i][j], false, false);
  };

  // ---- software-pipelined main loop (ping-pong LDS) ----
  constexpr int NK = K / KT;
  load_tile(0);
  store_tile(0);
  __syncthreads();
  #pragma unroll 2
  for (int kt = 0; kt < NK; ++kt) {
    if (kt + 1 < NK) load_tile((kt + 1) * KT);   // overlap with WMMAs below
    compute(kt & 1);
    if (kt + 1 < NK) store_tile((kt + 1) & 1);   // other buffer: no hazard
    __syncthreads();
  }

  // ---- epilogue: lane(lo)=N, rows m = v + 8*hi; immediate-offset stores ----
  #pragma unroll
  for (int i = 0; i < 2; ++i) {
    const int rbase = m0 + wm * 32 + i * 16 + hi * 8;
    #pragma unroll
    for (int j = 0; j < 4; ++j) {
      const int col = n0 + wn * 64 + j * 16 + lo;
      float bv = 0.f;
      if constexpr (EPI == EPI_TANH) bv = bias[col];
      float* p = Db + rbase * LDD + col;
      #pragma unroll
      for (int v = 0; v < 8; ++v) {
        float val = acc[i][j][v];
        if constexpr (EPI == EPI_TANH) val = tanhf(val + bv);
        p[v * LDD] = val;
      }
    }
  }
}

// Masked softmax over each [LK] row of attn, in place. One block per (q,b).
__global__ __launch_bounds__(256) void softmax_rows(float* __restrict__ attn,
                                                    const int* __restrict__ mask)
{
  const int q = blockIdx.x;
  const int b = blockIdx.y;
  float* row = attn + ((long long)b * LQ_ + q) * LK_;
  const int* mrow = mask + (long long)b * LK_;
  const int tid  = threadIdx.x;
  const int lane = tid & 31;
  const int w    = tid >> 5;
  __shared__ float red[8];

  float x[8];
  float mx = -1e30f;
  #pragma unroll
  for (int i = 0; i < 8; ++i) {
    int k = tid + i * 256;
    float v = row[k];
    if (mrow[k] != 0) v = -1e30f;   // masked out -> ~-inf
    x[i] = v;
    mx = fmaxf(mx, v);
  }
  #pragma unroll
  for (int off = 16; off > 0; off >>= 1)
    mx = fmaxf(mx, __shfl_xor(mx, off, 32));
  if (lane == 0) red[w] = mx;
  __syncthreads();
  mx = red[0];
  #pragma unroll
  for (int i = 1; i < 8; ++i) mx = fmaxf(mx, red[i]);
  __syncthreads();

  float sum = 0.f;
  #pragma unroll
  for (int i = 0; i < 8; ++i) {
    x[i] = __expf(x[i] - mx);
    sum += x[i];
  }
  #pragma unroll
  for (int off = 16; off > 0; off >>= 1)
    sum += __shfl_xor(sum, off, 32);
  if (lane == 0) red[w] = sum;
  __syncthreads();
  sum = 0.f;
  #pragma unroll
  for (int i = 0; i < 8; ++i) sum += red[i];
  const float inv = 1.0f / sum;
  #pragma unroll
  for (int i = 0; i < 8; ++i) row[tid + i * 256] = x[i] * inv;
}

extern "C" void kernel_launch(void* const* d_in, const int* in_sizes, int n_in,
                              void* d_out, int out_size, void* d_ws, size_t ws_size,
                              hipStream_t stream) {
  (void)in_sizes; (void)n_in; (void)out_size; (void)ws_size;
  const float* outp = (const float*)d_in[0];   // [B, LQ, H]
  const float* ctx  = (const float*)d_in[1];   // [B, LK, H]
  const int*   mask = (const int*)d_in[2];     // [B, LK]
  const float* W    = (const float*)d_in[3];   // [H, 2H]
  const float* bias = (const float*)d_in[4];   // [H]

  float* out_final = (float*)d_out;                         // [B, LQ, H]
  float* attn      = out_final + (long long)B_ * LQ_ * H_;  // [B, LQ, LK]
  float* mix       = (float*)d_ws;                          // [B, LQ, H] f32

  dim3 blk(256);

  // GEMM1: scores = O @ C^T   (K=H, LDA=H, LDB=H, LDD=LK)
  gemm_bf16_nt<EPI_STORE, false, false, H_, H_, H_, LK_>
      <<<dim3(LK_ / NTL, LQ_ / MT, B_), blk, 0, stream>>>(
      outp, nullptr, ctx, nullptr, attn,
      (long long)LQ_ * H_, (long long)LK_ * H_, (long long)LQ_ * LK_);

  // Masked softmax over k, in place in the attn output region
  softmax_rows<<<dim3(LQ_, B_), blk, 0, stream>>>(attn, mask);

  // GEMM2: mix = P @ C   (B source is [K=LK][N=H]; K=LK, LDA=LK, LDB=H, LDD=H)
  gemm_bf16_nt<EPI_STORE, true, false, LK_, LK_, H_, H_>
      <<<dim3(H_ / NTL, LQ_ / MT, B_), blk, 0, stream>>>(
      attn, nullptr, ctx, nullptr, mix,
      (long long)LQ_ * LK_, (long long)LK_ * H_, (long long)LQ_ * H_);

  // GEMM3: out = tanh([mix | O] @ W^T + b)   (K=2H, LDA=H, LDB=2H, LDD=H)
  gemm_bf16_nt<EPI_TANH, false, true, 2 * H_, H_, 2 * H_, H_>
      <<<dim3(H_ / NTL, LQ_ / MT, B_), blk, 0, stream>>>(
      mix, outp, W, bias, out_final,
      (long long)LQ_ * H_, 0LL, (long long)LQ_ * H_);
}
